// VarLSTMCell_22170621182346
// MI455X (gfx1250) — matrix operand order repaired and checked by reading, not compiled, not run on hardware
//
#include <hip/hip_runtime.h>

#define BATCH 16384
#define HDIM  512   // H == I == 512
#define EPS   1e-5f

typedef _Float16 v16h __attribute__((ext_vector_type(16)));
typedef _Float16 v8h  __attribute__((ext_vector_type(8)));
typedef _Float16 v4h  __attribute__((ext_vector_type(4)));
typedef float    v8f  __attribute__((ext_vector_type(8)));

__device__ __forceinline__ float sigmoidf_(float x) {
    return 1.0f / (1.0f + __expf(-x));
}

// ---------------------------------------------------------------------------
// One-time f32 -> f16 repack of W[4,512,512] and U[4,512,512] into workspace.
// Natural [g][h][i] layout already matches the WMMA B-operand fetch pattern
// (16 contiguous f16 along K=i per lane). 4 elements per thread, 8B stores.
// ---------------------------------------------------------------------------
__global__ void cvt_weights_kernel(const float* __restrict__ W,
                                   const float* __restrict__ U,
                                   _Float16* __restrict__ w16,
                                   _Float16* __restrict__ u16) {
    int i = (blockIdx.x * blockDim.x + threadIdx.x) * 4;
    float4 w = *(const float4*)(W + i);
    float4 u = *(const float4*)(U + i);
    v4h hw = {(_Float16)w.x, (_Float16)w.y, (_Float16)w.z, (_Float16)w.w};
    v4h hu = {(_Float16)u.x, (_Float16)u.y, (_Float16)u.z, (_Float16)u.w};
    *(v4h*)(w16 + i) = hw;
    *(v4h*)(u16 + i) = hu;
}

// ---------------------------------------------------------------------------
// Fused: 8 WMMA GEMMs + per-gate LayerNorm + LSTM pointwise + cell LayerNorm.
// Block = 512 threads (16 waves), owns 16 batch rows x all 512 columns.
// Wave w owns columns [w*32, w*32+32): 8 matrices x 2 col-tiles of 16x16.
// Software-pipelined: staging loads one K-step ahead; A/B fragments two
// matrices ahead, so L2 latency is covered by outstanding loads + WMMAs.
// ---------------------------------------------------------------------------
__global__ void __launch_bounds__(512, 1)
varlstm_fused_kernel(const float* __restrict__ input,
                     const float* __restrict__ hin,
                     const float* __restrict__ cin,
                     const float* __restrict__ mask_x,
                     const float* __restrict__ mask_h,
                     const _Float16* __restrict__ w16,
                     const _Float16* __restrict__ u16,
                     const float* __restrict__ bias_ih,
                     const float* __restrict__ bias_hh,
                     const float* __restrict__ ln_x_gamma,
                     const float* __restrict__ ln_x_beta,
                     const float* __restrict__ ln_h_gamma,
                     const float* __restrict__ ln_h_beta,
                     const float* __restrict__ ln_c_gamma,
                     const float* __restrict__ ln_c_beta,
                     float* __restrict__ out) {
    // LDS: staged A fragments (8 matrices, per-lane WMMA layout) + LN scratch
    __shared__ _Float16 ldsA[8 * 32 * 16];          // 8 KB
    __shared__ float2  part[8][16][16];             // 16 KB  (matrix, row, wave)
    __shared__ float2  stats2[8][16];               // 1 KB   (mean, rstd)
    __shared__ float2  cpart[16][16];               // 2 KB
    __shared__ float2  cstats2[16];                 // 128 B

    const int tid  = threadIdx.x;
    const int lane = tid & 31;
    const int wv   = tid >> 5;           // 0..15, column group
    const int half = lane >> 4;          // 0/1 -> rows 0-7 / 8-15 in C/D tiles
    const int col16 = lane & 15;
    const int rowBase = blockIdx.x << 4; // 16 batch rows per block

    // ---- staging role (invariant across K steps) ----
    const int sm    = tid >> 6;          // matrix 0..7 (0-3: x-side, 4-7: h-side)
    const int slane = (tid >> 1) & 31;   // destination fragment lane
    const int shalf = tid & 1;           // low/high 8 f16 of the fragment
    const int srow  = slane & 15;
    const int sgrp  = (slane >> 4) & 1;
    const float* dsrc;
    const float* msrc;
    if (sm < 4) {
        dsrc = input + (size_t)(rowBase + srow) * HDIM;
        msrc = mask_x + ((size_t)sm * BATCH + rowBase + srow) * HDIM;
    } else {
        dsrc = hin + (size_t)(rowBase + srow) * HDIM;
        msrc = mask_h + ((size_t)(sm - 4) * BATCH + rowBase + srow) * HDIM;
    }
    _Float16* aw = &ldsA[(sm * 32 + slane) * 16 + shalf * 8];
    const int kstag = sgrp * 16 + shalf * 8;   // K offset within 32-chunk

    // ---- B-operand base pointers per matrix (col-tile t adds 16*HDIM) ----
    const _Float16* bptr[8];
#pragma unroll
    for (int m = 0; m < 8; ++m) {
        const _Float16* base = (m < 4) ? (w16 + (size_t)m * HDIM * HDIM)
                                       : (u16 + (size_t)(m - 4) * HDIM * HDIM);
        int hcol = wv * 32 + col16;
        bptr[m] = base + (size_t)hcol * HDIM + half * 16;
    }

    v8f acc[8][2] = {};   // 8 matrices x 2 col-tiles, f32 16x16 accumulators

    // staging prefetch registers (loaded one K-step ahead)
    float4 sd0, sd1, sq0, sq1;
    {
        int k0 = kstag;    // kb = 0
        sd0 = *(const float4*)(dsrc + k0);
        sd1 = *(const float4*)(dsrc + k0 + 4);
        sq0 = *(const float4*)(msrc + k0);
        sq1 = *(const float4*)(msrc + k0 + 4);
    }

    // =================== K loop: 512 / 32 = 16 steps ===================
    for (int kb = 0; kb < 16; ++kb) {
        __syncthreads();   // previous iteration's LDS reads complete
        // --- convert prefetched masked A data to f16, store to LDS ---
        {
            v8h hv;
            hv[0] = (_Float16)(sd0.x * sq0.x); hv[1] = (_Float16)(sd0.y * sq0.y);
            hv[2] = (_Float16)(sd0.z * sq0.z); hv[3] = (_Float16)(sd0.w * sq0.w);
            hv[4] = (_Float16)(sd1.x * sq1.x); hv[5] = (_Float16)(sd1.y * sq1.y);
            hv[6] = (_Float16)(sd1.z * sq1.z); hv[7] = (_Float16)(sd1.w * sq1.w);
            *(v8h*)aw = hv;
        }
        // --- issue staging loads for kb+1 (consumed next iteration) ---
        if (kb < 15) {
            int k0 = (kb + 1) * 32 + kstag;
            sd0 = *(const float4*)(dsrc + k0);
            sd1 = *(const float4*)(dsrc + k0 + 4);
            sq0 = *(const float4*)(msrc + k0);
            sq1 = *(const float4*)(msrc + k0 + 4);
        }
        // --- issue B loads for m=0,1 before the barrier (global only) ---
        v16h a_[8], b0_[8], b1_[8];
        b0_[0] = *(const v16h*)(bptr[0] + kb * 32);
        b1_[0] = *(const v16h*)(bptr[0] + (size_t)16 * HDIM + kb * 32);
        b0_[1] = *(const v16h*)(bptr[1] + kb * 32);
        b1_[1] = *(const v16h*)(bptr[1] + (size_t)16 * HDIM + kb * 32);
        __syncthreads();   // staged A fragments visible
        a_[0] = *(const v16h*)(ldsA + (0 * 32 + lane) * 16);
        a_[1] = *(const v16h*)(ldsA + (1 * 32 + lane) * 16);
        // --- pipelined WMMA over 8 matrices (prefetch depth 2) ---
#pragma unroll
        for (int m = 0; m < 8; ++m) {
            if (m + 2 < 8) {
                b0_[m + 2] = *(const v16h*)(bptr[m + 2] + kb * 32);
                b1_[m + 2] = *(const v16h*)(bptr[m + 2] + (size_t)16 * HDIM + kb * 32);
                a_[m + 2]  = *(const v16h*)(ldsA + ((m + 2) * 32 + lane) * 16);
            }
            acc[m][0] = __builtin_amdgcn_wmma_f32_16x16x32_f16(
                false, a_[m], false, b0_[m], (short)0, acc[m][0], false, false);
            acc[m][1] = __builtin_amdgcn_wmma_f32_16x16x32_f16(
                false, a_[m], false, b1_[m], (short)0, acc[m][1], false, false);
        }
    }

    // =================== per-gate LayerNorm statistics ===================
#pragma unroll
    for (int m = 0; m < 8; ++m) {
#pragma unroll
        for (int r = 0; r < 8; ++r) {
            float s = acc[m][0][r] + acc[m][1][r];
            float q = acc[m][0][r] * acc[m][0][r] + acc[m][1][r] * acc[m][1][r];
#pragma unroll
            for (int d = 1; d < 16; d <<= 1) {
                s += __shfl_xor(s, d, 32);
                q += __shfl_xor(q, d, 32);
            }
            if (col16 == 0) part[m][half * 8 + r][wv] = make_float2(s, q);
        }
    }
    __syncthreads();
    if (tid < 128) {     // one thread per (matrix,row): reduce 16 wave partials
        int m = tid >> 4, row = tid & 15;
        float s = 0.f, q = 0.f;
        for (int ww = 0; ww < 16; ++ww) {
            float2 p = part[m][row][ww];
            s += p.x; q += p.y;
        }
        float mu  = s * (1.0f / HDIM);
        float var = q * (1.0f / HDIM) - mu * mu;
        stats2[m][row] = make_float2(mu, rsqrtf(var + EPS));
    }
    __syncthreads();

    // =================== normalize + gates + cell update ===================
#pragma unroll
    for (int t = 0; t < 2; ++t) {
        int col = wv * 32 + t * 16 + col16;
        float lxg[4], lxb[4], lhg[4], lhb[4], bi[4], bh[4];
#pragma unroll
        for (int g = 0; g < 4; ++g) {
            lxg[g] = ln_x_gamma[g * HDIM + col];
            lxb[g] = ln_x_beta [g * HDIM + col];
            lhg[g] = ln_h_gamma[g * HDIM + col];
            lhb[g] = ln_h_beta [g * HDIM + col];
            bi [g] = bias_ih   [g * HDIM + col];
            bh [g] = bias_hh   [g * HDIM + col];
        }
#pragma unroll
        for (int r = 0; r < 8; ++r) {
            int row = half * 8 + r;
            float pre[4];
#pragma unroll
            for (int g = 0; g < 4; ++g) {
                float2 sx = stats2[g][row];
                float2 sh = stats2[g + 4][row];
                float xn = (acc[g][t][r]     - sx.x) * sx.y * lxg[g] + lxb[g];
                float hn = (acc[g + 4][t][r] - sh.x) * sh.y * lhg[g] + lhb[g];
                pre[g] = xn + hn + bi[g] + bh[g];
            }
            float it = sigmoidf_(pre[0]);
            float ft = sigmoidf_(pre[1]);
            float gt = tanhf(pre[2]);
            float ot = sigmoidf_(pre[3]);
            float cv = cin[(size_t)(rowBase + row) * HDIM + col];
            float ct = ft * cv + it * gt;
            acc[0][t][r] = ct;   // stash c_t
            acc[1][t][r] = ot;   // stash o_t
        }
    }

    // =================== cell LayerNorm statistics ===================
#pragma unroll
    for (int r = 0; r < 8; ++r) {
        float c0 = acc[0][0][r], c1 = acc[0][1][r];
        float s = c0 + c1, q = c0 * c0 + c1 * c1;
#pragma unroll
        for (int d = 1; d < 16; d <<= 1) {
            s += __shfl_xor(s, d, 32);
            q += __shfl_xor(q, d, 32);
        }
        if (col16 == 0) cpart[half * 8 + r][wv] = make_float2(s, q);
    }
    __syncthreads();
    if (tid < 16) {
        float s = 0.f, q = 0.f;
        for (int ww = 0; ww < 16; ++ww) {
            float2 p = cpart[tid][ww];
            s += p.x; q += p.y;
        }
        float mu  = s * (1.0f / HDIM);
        float var = q * (1.0f / HDIM) - mu * mu;
        cstats2[tid] = make_float2(mu, rsqrtf(var + EPS));
    }
    __syncthreads();

    // =================== h_t = o * tanh(LN(c_t)); store h_t, c_t ===========
#pragma unroll
    for (int t = 0; t < 2; ++t) {
        int col = wv * 32 + t * 16 + col16;
        float lcg = ln_c_gamma[col];
        float lcb = ln_c_beta[col];
#pragma unroll
        for (int r = 0; r < 8; ++r) {
            int row = half * 8 + r;
            float2 sc = cstats2[row];
            float ct = acc[0][t][r];
            float ot = acc[1][t][r];
            float ht = ot * tanhf((ct - sc.x) * sc.y * lcg + lcb);
            size_t off = (size_t)(rowBase + row) * HDIM + col;
            out[off] = ht;                                // h_t
            out[(size_t)BATCH * HDIM + off] = ct;         // c_t
        }
    }
}

extern "C" void kernel_launch(void* const* d_in, const int* in_sizes, int n_in,
                              void* d_out, int out_size, void* d_ws, size_t ws_size,
                              hipStream_t stream) {
    const float* input      = (const float*)d_in[0];
    const float* h          = (const float*)d_in[1];
    const float* c          = (const float*)d_in[2];
    const float* mask_x     = (const float*)d_in[3];
    const float* mask_h     = (const float*)d_in[4];
    const float* W          = (const float*)d_in[5];
    const float* U          = (const float*)d_in[6];
    const float* bias_ih    = (const float*)d_in[7];
    const float* bias_hh    = (const float*)d_in[8];
    const float* ln_x_gamma = (const float*)d_in[9];
    const float* ln_x_beta  = (const float*)d_in[10];
    const float* ln_h_gamma = (const float*)d_in[11];
    const float* ln_h_beta  = (const float*)d_in[12];
    const float* ln_c_gamma = (const float*)d_in[13];
    const float* ln_c_beta  = (const float*)d_in[14];

    _Float16* w16 = (_Float16*)d_ws;                    // 2 MB
    _Float16* u16 = w16 + (size_t)4 * HDIM * HDIM;      // 2 MB  (4 MB ws total)

    // 4*512*512 = 1,048,576 elems per tensor; 4 per thread -> 1024 blocks x 256
    cvt_weights_kernel<<<1024, 256, 0, stream>>>(W, U, w16, u16);

    // 16384 / 16 rows per block = 1024 blocks of 512 threads (16 waves)
    varlstm_fused_kernel<<<BATCH / 16, 512, 0, stream>>>(
        input, h, c, mask_x, mask_h, w16, u16, bias_ih, bias_hh,
        ln_x_gamma, ln_x_beta, ln_h_gamma, ln_h_beta, ln_c_gamma, ln_c_beta,
        (float*)d_out);
}
